// model_GRU_classifier_58402965291683
// MI455X (gfx1250) — compile-verified, matrix operand order checked by reading, not run on hardware
//
#include <hip/hip_runtime.h>

typedef __attribute__((ext_vector_type(8)))  _Float16 v8h;
typedef __attribute__((ext_vector_type(16))) _Float16 v16h;
typedef __attribute__((ext_vector_type(8)))  float    v8f;
typedef __attribute__((ext_vector_type(4)))  unsigned int u32x4;
typedef __attribute__((ext_vector_type(8)))  int      i32x8;
typedef __attribute__((ext_vector_type(4)))  int      i32x4;

#define B_  64
#define T_  1024
#define E_  768     // n_embd
#define H_  256     // hidden
#define G3_ 768     // 3*H
#define C_  2

// Build a 16-element f16 WMMA fragment from two 16B-aligned 8-element chunks.
__device__ __forceinline__ v16h ld_frag(const _Float16* p0, const _Float16* p1) {
    v8h lo = *(const v8h*)p0;
    v8h hi = *(const v8h*)p1;
    return __builtin_shufflevector(lo, hi, 0,1,2,3,4,5,6,7,8,9,10,11,12,13,14,15);
}

// ---------------------------------------------------------------------------
// TDM: async 2D tile load (16 rows x 768 f16, row stride T_*G3_ elements)
// from global x_proj into LDS.  Descriptor packed per CDNA5 ISA §8.3/8.4.
// This toolchain exposes the 6-arg builtin (extra i32x8 group, zero-filled).
// ---------------------------------------------------------------------------
__device__ __forceinline__ void tdm_load_xp_tile(const _Float16* gsrc,
                                                 unsigned lds_off_bytes) {
    unsigned long long ga = (unsigned long long)gsrc;
    const unsigned long long stride0 = (unsigned long long)T_ * G3_; // elems

    u32x4 g0;
    g0[0] = 1u;                                   // count=1 valid, user mode
    g0[1] = lds_off_bytes;                        // lds_addr (bytes)
    g0[2] = (unsigned)(ga & 0xffffffffull);       // global_addr[31:0]
    g0[3] = (unsigned)((ga >> 32) & 0x1ffffffull) // global_addr[56:32]
          | (2u << 30);                           // type = 2 ("image")

    i32x8 g1;
    g1[0] = (int)(1u << 16);                      // data_size=1 (2B); wg_mask=0
    g1[1] = (int)((unsigned)(G3_ & 0xffff) << 16);          // tensor_dim0 lo
    g1[2] = (int)((G3_ >> 16) | (16u << 16));               // dim0 hi | tensor_dim1=16
    g1[3] = (int)((unsigned)G3_ << 16);                     // tile_dim0 = 768
    g1[4] = 16;                                             // tile_dim1=16, tile_dim2=0
    g1[5] = (int)(unsigned)(stride0 & 0xffffffffull);       // dim0_stride[31:0]
    g1[6] = (int)(unsigned)((stride0 >> 32) & 0xffff);      // dim0_stride[47:32]
    g1[7] = 0;                                              // dim1_stride (unused)

    i32x4 gz4 = {0, 0, 0, 0};                     // 2D tensor: groups 2/3 zero
    i32x8 gz8 = {0, 0, 0, 0, 0, 0, 0, 0};         // extra group (6-arg form)
    __builtin_amdgcn_tensor_load_to_lds(g0, g1, gz4, gz4, gz8, 0);
}

// ---------------------------------------------------------------------------
// Kernel 1: convert weights to f16 (w_ih, w_hh) and build zero-padded w_score
// ---------------------------------------------------------------------------
__global__ __launch_bounds__(256) void k_convert(
    const float* __restrict__ wih, const float* __restrict__ whh,
    const float* __restrict__ wsc,
    _Float16* __restrict__ wih_h, _Float16* __restrict__ whh_h,
    _Float16* __restrict__ wsc_h) {
    int i = blockIdx.x * 256 + threadIdx.x;
    if (i < G3_ * E_) wih_h[i] = (_Float16)wih[i];
    if (i < G3_ * H_) whh_h[i] = (_Float16)whh[i];
    if (i < 16 * H_) {
        int n = i >> 8, k = i & (H_ - 1);
        wsc_h[i] = (_Float16)((n < C_) ? wsc[n * H_ + k] : 0.0f);
    }
}

// ---------------------------------------------------------------------------
// Kernel 2: x_proj[mg, 0:768] = f16( emb[seq[mg]] @ w_ih^T + b_ih )
// Embedding gather via global_load_async_to_lds_b128 (ASYNCcnt path),
// then a wave-local f32->f16 convert into the WMMA A tile.
// ---------------------------------------------------------------------------
__global__ __launch_bounds__(256) void k_xproj(
    const int* __restrict__ seq, const float* __restrict__ emb,
    const _Float16* __restrict__ wih, const float* __restrict__ bih,
    _Float16* __restrict__ xp) {
    __shared__ __align__(16) float    sF[16][E_];       // raw f32 rows (async dst)
    __shared__ __align__(16) _Float16 sA[16][E_ + 8];   // f16 A tile (padded)

    const int tid  = threadIdx.x;
    const int lane = tid & 31, wave = tid >> 5;
    const int half = lane >> 4, ln = lane & 15;
    const long mg0 = (long)blockIdx.x * 16;

    // ---- async gather: 12 x 16B per thread, memory -> LDS (no VGPR data) ----
    {
        int m  = tid >> 4;
        int c0 = (tid & 15) * 48;
        long row = (long)seq[mg0 + m] * E_ + c0;
        unsigned lds0 = (unsigned)(unsigned long long)&sF[m][c0];
        unsigned long long ga = (unsigned long long)(emb + row);
        #pragma unroll
        for (int c = 0; c < 12; ++c) {
            unsigned lo = lds0 + c * 16;
            unsigned long long g = ga + (unsigned long long)c * 16;
            asm volatile("global_load_async_to_lds_b128 %0, %1, off"
                         :: "v"(lo), "v"(g) : "memory");
        }
        asm volatile("s_wait_asynccnt 0x0" ::: "memory");
        for (int c = 0; c < 48; ++c)                 // wave-local convert
            sA[m][c0 + c] = (_Float16)sF[m][c0 + c];
    }
    __syncthreads();

    const int nb0 = wave * 6;                        // 48 N-tiles / 8 waves
    v8f acc[6];
    #pragma unroll
    for (int i = 0; i < 6; ++i) { v8f z = {}; acc[i] = z; }

    for (int kb = 0; kb < E_; kb += 32) {
        v16h a = ld_frag(&sA[ln][kb + half * 8], &sA[ln][kb + 16 + half * 8]);
        #pragma unroll
        for (int i = 0; i < 6; ++i) {
            int n = (nb0 + i) * 16 + ln;             // B column n = row n of w_ih
            const _Float16* bp = wih + (long)n * E_ + kb + half * 8;
            v16h bf = ld_frag(bp, bp + 16);
            acc[i] = __builtin_amdgcn_wmma_f32_16x16x32_f16(
                false, a, false, bf, (short)0, acc[i], false, false);
        }
    }

    #pragma unroll
    for (int i = 0; i < 6; ++i) {
        int n = (nb0 + i) * 16 + ln;
        float bias = bih[n];
        #pragma unroll
        for (int r = 0; r < 8; ++r) {
            long mg = mg0 + r + 8 * half;            // C frag: M = r + 8*(lane>=16)
            xp[mg * G3_ + n] = (_Float16)(acc[i][r] + bias);
        }
    }
}

// ---------------------------------------------------------------------------
// Kernel 3: GRU scan + fused score.  4 workgroups x 16 batch rows.
// x_proj tiles for step t+1 are TDM-prefetched into double-buffered LDS
// while step t's recurrent WMMA runs; synced with s_wait_tensorcnt.
// ---------------------------------------------------------------------------
__global__ __launch_bounds__(256) void k_gru(
    const _Float16* __restrict__ xp, const _Float16* __restrict__ whh,
    const float* __restrict__ bhh, const _Float16* __restrict__ wsc,
    float* __restrict__ out) {
    __shared__ __align__(16) _Float16 hA[16][H_ + 8];    // h (f16)  : WMMA A
    __shared__ __align__(16) _Float16 hR[16][H_ + 8];    // relu(h)  : score A
    __shared__ __align__(16) _Float16 hp[16][G3_ + 8];   // gate pre-activations
    __shared__ __align__(16) _Float16 sX[2][16][G3_];    // TDM-staged x_proj

    const int tid  = threadIdx.x;
    const int lane = tid & 31, wave = tid >> 5;
    const int half = lane >> 4, ln = lane & 15;
    const int bbase = blockIdx.x * 16;

    const int nb0 = wave * 6;
    float bh[6];
    #pragma unroll
    for (int i = 0; i < 6; ++i) {
        int n = (nb0 + i) * 16 + ln;
        bh[i] = bhh[n];
        __builtin_prefetch(whh + (long)n * H_, 0, 3);    // warm L2/L0 path
    }

    // per-thread h state: row em, 16 hidden elems starting at j0
    const int em = tid >> 4;
    const int j0 = (tid & 15) * 16;
    float h[16];
    #pragma unroll
    for (int jj = 0; jj < 16; ++jj) {
        h[jj] = 0.0f;
        hA[em][j0 + jj] = (_Float16)0.0f;
    }

    // prologue: TDM-load x_proj tile for t=0 into buffer 0
    if (wave == 0) {
        tdm_load_xp_tile(xp + ((long)bbase * T_ + 0) * G3_,
                         (unsigned)(unsigned long long)&sX[0][0][0]);
        __builtin_amdgcn_s_wait_tensorcnt(0);
    }
    __syncthreads();

    for (int t = 0; t < T_; ++t) {
        // kick off next step's x_proj tile DMA (overlaps with WMMA below)
        if (wave == 0 && (t + 1) < T_) {
            tdm_load_xp_tile(xp + ((long)bbase * T_ + (t + 1)) * G3_,
                             (unsigned)(unsigned long long)&sX[(t + 1) & 1][0][0]);
        }

        // ---- phase 1: hp = h @ w_hh^T + b_hh (WMMA) ----
        v8f acc[6];
        #pragma unroll
        for (int i = 0; i < 6; ++i) { v8f z = {}; acc[i] = z; }
        for (int kb = 0; kb < H_; kb += 32) {
            v16h a = ld_frag(&hA[ln][kb + half * 8], &hA[ln][kb + 16 + half * 8]);
            #pragma unroll
            for (int i = 0; i < 6; ++i) {
                int n = (nb0 + i) * 16 + ln;
                const _Float16* bp = whh + (long)n * H_ + kb + half * 8;
                v16h bf = ld_frag(bp, bp + 16);
                acc[i] = __builtin_amdgcn_wmma_f32_16x16x32_f16(
                    false, a, false, bf, (short)0, acc[i], false, false);
            }
        }
        #pragma unroll
        for (int i = 0; i < 6; ++i) {
            int n = (nb0 + i) * 16 + ln;
            #pragma unroll
            for (int r = 0; r < 8; ++r)
                hp[r + 8 * half][n] = (_Float16)(acc[i][r] + bh[i]);
        }
        if (wave == 0) __builtin_amdgcn_s_wait_tensorcnt(0);  // t+1 tile landed
        __syncthreads();

        // ---- phase 2: gates + state update (h stays in registers) ----
        const _Float16* xrow = &sX[t & 1][em][0];
        #pragma unroll
        for (int jj = 0; jj < 16; ++jj) {
            int j = j0 + jj;
            float xr = (float)xrow[j];
            float xz = (float)xrow[H_ + j];
            float xn = (float)xrow[2 * H_ + j];
            float hr = (float)hp[em][j];
            float hz = (float)hp[em][H_ + j];
            float hn = (float)hp[em][2 * H_ + j];
            float rg = 1.0f / (1.0f + __expf(-(xr + hr)));
            float zg = 1.0f / (1.0f + __expf(-(xz + hz)));
            float e2 = __expf(2.0f * (xn + rg * hn));
            float ng = 1.0f - 2.0f / (e2 + 1.0f);         // tanh, inf-safe
            float hv = (1.0f - zg) * ng + zg * h[jj];
            h[jj] = hv;
            hA[em][j] = (_Float16)hv;
            hR[em][j] = (_Float16)(hv > 0.0f ? hv : 0.0f);
        }
        __syncthreads();

        // ---- phase 3: out[b, t, :] = relu(h) @ w_score^T (wave 0) ----
        if (wave == 0) {
            v8f sc = {};
            for (int kb = 0; kb < H_; kb += 32) {
                v16h a = ld_frag(&hR[ln][kb + half * 8], &hR[ln][kb + 16 + half * 8]);
                const _Float16* bp = wsc + (long)ln * H_ + kb + half * 8;
                v16h bf = ld_frag(bp, bp + 16);
                sc = __builtin_amdgcn_wmma_f32_16x16x32_f16(
                    false, a, false, bf, (short)0, sc, false, false);
            }
            if (ln < C_) {
                #pragma unroll
                for (int r = 0; r < 8; ++r) {
                    int bb = bbase + r + 8 * half;
                    out[((long)bb * T_ + t) * C_ + ln] = sc[r];
                }
            }
        }
        __syncthreads();
    }
}

// ---------------------------------------------------------------------------
extern "C" void kernel_launch(void* const* d_in, const int* in_sizes, int n_in,
                              void* d_out, int out_size, void* d_ws, size_t ws_size,
                              hipStream_t stream) {
    const int*   seq = (const int*)d_in[0];
    const float* emb = (const float*)d_in[1];
    const float* wih = (const float*)d_in[2];
    const float* whh = (const float*)d_in[3];
    const float* bih = (const float*)d_in[4];
    const float* bhh = (const float*)d_in[5];
    const float* wsc = (const float*)d_in[6];
    float* out = (float*)d_out;

    // workspace layout (bytes), all 16B-aligned
    char* ws = (char*)d_ws;
    _Float16* wih_h = (_Float16*)(ws);                           // 768*768
    _Float16* whh_h = (_Float16*)(ws + 1179648);                 // 768*256
    _Float16* wsc_h = (_Float16*)(ws + 1179648 + 393216);        // 16*256 padded
    _Float16* xp    = (_Float16*)(ws + 1179648 + 393216 + 8192); // 65536*768 f16

    k_convert<<<(G3_ * E_ + 255) / 256, 256, 0, stream>>>(wih, whh, wsc,
                                                          wih_h, whh_h, wsc_h);
    k_xproj<<<(B_ * T_) / 16, 256, 0, stream>>>(seq, emb, wih_h, bih, xp);
    k_gru<<<B_ / 16, 256, 0, stream>>>(xp, whh_h, bhh, wsc_h, out);
}
